// Broyden_79998060855493
// MI455X (gfx1250) — compile-verified, matrix operand order checked by reading, not run on hardware
//
#include <hip/hip_runtime.h>
#include <stdint.h>

// ---------------- problem constants (from reference) ----------------
#define BB    16
#define CIN   64
#define COUT  128
#define HH    32
#define WW    32
#define DD    (COUT*HH*WW)      // 131072 per-batch vector length
#define NT    (BB*DD)           // 2097152 total vector elems
#define KMAX  20
#define NPOS  (BB*HH*WW)        // 16384 GEMM columns
#define PADP  (34*34)           // padded plane size

typedef __attribute__((ext_vector_type(16))) __bf16 v16bf;
typedef __attribute__((ext_vector_type(8)))  float  v8f;

// ---------------- gfx1250 async global->LDS support (guarded) ----------------
#if defined(__gfx1250__) && __has_builtin(__builtin_amdgcn_global_load_async_to_lds_b128)
#define ASYNC_LDS 1
#else
#define ASYNC_LDS 0
#endif

// payload type exactly as the builtin expects: 16-byte int vector
typedef int v4i_vs __attribute__((__vector_size__(4 * sizeof(int))));
typedef __attribute__((address_space(1))) v4i_vs* g_v4i_ptr;
typedef __attribute__((address_space(3))) v4i_vs* l_v4i_ptr;

__device__ __forceinline__ void async_copy_b128(const uint16_t* g, uint16_t* l) {
#if ASYNC_LDS
    // generic LDS address low 32 bits == LDS byte offset (ISA aperture rule)
    __builtin_amdgcn_global_load_async_to_lds_b128(
        (g_v4i_ptr)(uintptr_t)g,
        (l_v4i_ptr)(uint32_t)(uintptr_t)l,
        0, 0);
#endif
}
__device__ __forceinline__ void async_wait_all() {
#if ASYNC_LDS
#if __has_builtin(__builtin_amdgcn_s_wait_asynccnt)
    __builtin_amdgcn_s_wait_asynccnt(0);
#else
    asm volatile("s_wait_asynccnt 0" ::: "memory");
#endif
#endif
}

// ---------------- helpers ----------------
__device__ __forceinline__ uint16_t f2bf(float f) {
    union { float f; uint32_t u; } c; c.f = f;
    uint32_t u = c.u;
    uint32_t r = (u + 0x7FFFu + ((u >> 16) & 1u)) >> 16;
    return (uint16_t)r;
}

__device__ __forceinline__ v16bf load_frag32(const uint16_t* p0, const uint16_t* p1) {
    union { struct { uint4 a, b; } u; v16bf v; } c;
    c.u.a = *(const uint4*)p0;
    c.u.b = *(const uint4*)p1;
    return c.v;
}

__device__ __forceinline__ float block_reduce_256(float v) {
    __shared__ float red[256];
    int t = threadIdx.x;
    red[t] = v; __syncthreads();
    #pragma unroll
    for (int s = 128; s > 0; s >>= 1) {
        if (t < s) red[t] += red[t + s];
        __syncthreads();
    }
    return red[0];
}

// ---------------- weight packing into per-lane WMMA fragment order ----------------
// Wp[((chunk*8 + mtile)*32 + lane)*16 + i] , GEMM k = r*Cin + ci (r = kh*3+kw)
__global__ void pack_w_frag_kernel(const float* __restrict__ W, int Cin,
                                   uint16_t* __restrict__ Wp) {
    int idx = blockIdx.x * 256 + threadIdx.x;
    int total = Cin * 9 * COUT;
    if (idx >= total) return;
    int i    = idx & 15;
    int lane = (idx >> 4) & 31;
    int mt   = (idx >> 9) & 7;
    int c    = idx >> 12;
    int half = lane >> 4;
    int m    = mt * 16 + (lane & 15);                 // output channel
    int kk   = c * 32 + ((i < 8) ? (half * 8 + i) : (16 + half * 8 + (i - 8)));
    int r    = kk / Cin;
    int ci   = kk % Cin;
    Wp[idx] = f2bf(W[((size_t)m * Cin + ci) * 9 + r]);
}

// ---------------- fp32 [n][C][32][32] -> bf16 padded [n][C][34][34] ----------------
__global__ void pad_cvt_kernel(const float* __restrict__ s, uint16_t* __restrict__ zp) {
    int i = blockIdx.x * 256 + threadIdx.x;
    int w = i & 31, h = (i >> 5) & 31, c = i >> 10;   // c = n*C + ci
    zp[((size_t)c * 34 + (h + 1)) * 34 + (w + 1)] = f2bf(s[i]);
}

// fused: xo = a + bvec ; zp(padded bf16) = bf16(xo)
__global__ void vadd_pad_kernel(const float* __restrict__ a, const float* __restrict__ bvec,
                                float* __restrict__ xo, uint16_t* __restrict__ zp) {
    int i = blockIdx.x * 256 + threadIdx.x;
    float v = a[i] + bvec[i];
    xo[i] = v;
    int w = i & 31, h = (i >> 5) & 31, c = i >> 10;
    zp[((size_t)c * 34 + (h + 1)) * 34 + (w + 1)] = f2bf(v);
}

// ---------------- implicit-GEMM 3x3 SAME conv via WMMA bf16 ----------------
// Block: 8 waves, tile = 128 co x 64 columns, double-buffered LDS staging
// (async weight pipeline on gfx1250), 4 WMMA per wave per k-step.
// MODE 0: out = y + bias[co]
// MODE 1: out = relu(y+addv[i]) - xv[i]
// MODE 2: out = relu(y+addv[i])
// MODE 3: g = relu(y+addv[i]) - xv[i]; out = g; out2 = g - gxold[i]
template<int CINT, int MODE>
__global__ __launch_bounds__(256)
void conv_wmma_kernel(const uint16_t* __restrict__ Zp,
                      const uint16_t* __restrict__ Wp,
                      const float* __restrict__ addv,
                      const float* __restrict__ xv,
                      const float* __restrict__ gxold,
                      float* __restrict__ out,
                      float* __restrict__ out2) {
    constexpr int KTOT  = CINT * 9;
    constexpr int NSTEP = KTOT / 32;
    __shared__ __align__(16) uint16_t Zs[2][64 * 40];  // [j][k], 80B rows -> aligned b128 reads
#if ASYNC_LDS
    __shared__ __align__(16) uint16_t Wl[2][4096];     // double-buffered 8KB weight slabs
#endif

    const int tid  = threadIdx.x;
    const int wave = tid >> 5;
    const int lane = tid & 31;
    const int half = lane >> 4, l15 = lane & 15;
    const int pbase = blockIdx.x * 64;
    const int n  = pbase >> 10;
    const int h0 = (pbase >> 5) & 31;                 // rows h0, h0+1 (h0 even)

    // staging map: thread -> (ci-pair cp, j-quartet jq)
    const int jq = tid >> 4;                          // j = jq*4 + i
    const int cp = tid & 15;                          // ci pair base = 2*cp

    auto stage = [&](int step, int buf) {
        const int kb  = step * 32;
        const int r   = kb / CINT;                    // compile-time strength-reduced
        const int cib = kb % CINT;
        const int dh  = r / 3 - 1;
        const int dw  = r % 3 - 1;
        const int j0  = jq * 4;
        const int h2  = h0 + (j0 >> 5) + dh + 1;      // padded coords, in-bounds
        const int w2  = (j0 & 31) + dw + 1;
        const uint16_t* src0 = Zp + (((size_t)(n * CINT + cib + 2 * cp)) * 34 + h2) * 34 + w2;
        const uint16_t* src1 = src0 + PADP;
        #pragma unroll
        for (int i2 = 0; i2 < 4; ++i2) {
            uint32_t pv = (uint32_t)src0[i2] | ((uint32_t)src1[i2] << 16);
            *(uint32_t*)&Zs[buf][(size_t)(j0 + i2) * 40 + 2 * cp] = pv;  // conflict-free b32
        }
    };

    auto stage_w = [&](int step, int buf) {
#if ASYNC_LDS
        const uint16_t* g = Wp + (size_t)step * 4096 + tid * 16;   // contiguous 8KB slab
        async_copy_b128(g,     &Wl[buf][tid * 16]);
        async_copy_b128(g + 8, &Wl[buf][tid * 16 + 8]);
#else
        (void)step; (void)buf;
#endif
    };

    v8f acc[4] = {v8f{}, v8f{}, v8f{}, v8f{}};

    stage(0, 0);
    stage_w(0, 0);
    #pragma unroll 2
    for (int s = 0; s < NSTEP; ++s) {
        async_wait_all();                              // asyncs issued last iteration
        __syncthreads();                               // buffers s&1 ready
        if (s + 1 < NSTEP) {                           // overlap next stage with math
            stage(s + 1, (s + 1) & 1);
            stage_w(s + 1, (s + 1) & 1);
        }

#if ASYNC_LDS
        const uint16_t* ap = &Wl[s & 1][tid * 16];     // == (wave*32+lane)*16
        v16bf af = load_frag32(ap, ap + 8);
#else
        const uint16_t* ap = Wp + (((size_t)s * 8 + wave) * 32 + lane) * 16;
        v16bf af = load_frag32(ap, ap + 8);
        __builtin_prefetch((const void*)(ap + 4096), 0, 1);
#endif

        const uint16_t* zb = Zs[s & 1];
        #pragma unroll
        for (int jt = 0; jt < 4; ++jt) {
            const uint16_t* rowb = zb + (size_t)(jt * 16 + l15) * 40;
            v16bf bfrag = load_frag32(rowb + half * 8, rowb + 16 + half * 8);
            acc[jt] = __builtin_amdgcn_wmma_f32_16x16x32_bf16(false, af, false, bfrag,
                                                              (short)0, acc[jt], false, false);
        }
    }

    // epilogue: D layout lane L: N=l15, VGPR v -> M = v + 8*half
    #pragma unroll
    for (int jt = 0; jt < 4; ++jt) {
        const int j = jt * 16 + l15;
        const int h = h0 + (j >> 5);
        const int w = j & 31;
        #pragma unroll
        for (int vv = 0; vv < 8; ++vv) {
            int co = wave * 16 + half * 8 + vv;
            size_t oi = (((size_t)n * COUT + co) * HH + h) * WW + w;
            float y = acc[jt][vv];
            if (MODE == 0) {
                out[oi] = y + addv[co];
            } else if (MODE == 1) {
                float t = y + addv[oi];
                t = t > 0.f ? t : 0.f;
                out[oi] = t - xv[oi];
            } else if (MODE == 2) {
                float t = y + addv[oi];
                out[oi] = t > 0.f ? t : 0.f;
            } else {
                float t = y + addv[oi];
                t = t > 0.f ? t : 0.f;
                float g = t - xv[oi];
                out[oi]  = g;
                out2[oi] = g - gxold[oi];
            }
        }
    }
}

// t[b,j] = sum_d v[b,d] * Us[b,d,j]   (grid = B*kcur)
__global__ void colsdot_kernel(const float* __restrict__ v, const float* __restrict__ Us,
                               float* __restrict__ outv, int kcur) {
    int b = blockIdx.x / kcur, j = blockIdx.x % kcur;
    const float* vb = v + (size_t)b * DD;
    const float* Mb = Us + (size_t)b * DD * KMAX + j;
    float s = 0.f;
    for (int d = threadIdx.x; d < DD; d += 256) s += vb[d] * Mb[(size_t)d * KMAX];
    s = block_reduce_256(s);
    if (threadIdx.x == 0) outv[b * KMAX + j] = s;
}

// t[b,j] = sum_d VTs[b,j,d] * v[b,d]   (coalesced; grid = B*kcur)
__global__ void rowsdot_kernel(const float* __restrict__ v, const float* __restrict__ VTs,
                               float* __restrict__ outv, int kcur) {
    int b = blockIdx.x / kcur, j = blockIdx.x % kcur;
    const float* vb = v + (size_t)b * DD;
    const float* Mb = VTs + ((size_t)b * KMAX + j) * DD;
    float s = 0.f;
    for (int d = threadIdx.x; d < DD; d += 256) s += vb[d] * Mb[d];
    s = block_reduce_256(s);
    if (threadIdx.x == 0) outv[b * KMAX + j] = s;
}

// o[b,d] = -dx[b,d] + sum_j coef[b,j] * VTs[b,j,d]
__global__ void combine_rows_kernel(const float* __restrict__ dx, const float* __restrict__ coef,
                                    const float* __restrict__ VTs, float* __restrict__ o, int kcur) {
    int i = blockIdx.x * 256 + threadIdx.x;
    int b = i / DD, d = i % DD;
    __shared__ float cs[KMAX];
    if (threadIdx.x < kcur) cs[threadIdx.x] = coef[b * KMAX + threadIdx.x];
    __syncthreads();
    float s = -dx[i];
    for (int j = 0; j < kcur; ++j) s += cs[j] * VTs[((size_t)b * KMAX + j) * DD + d];
    o[i] = s;
}

// o[i] = sgn * ( sum_j Us[b,d,j]*coef[b,j] - v[i] )
__global__ void lowrank_mv_kernel(const float* __restrict__ v, const float* __restrict__ Us,
                                  const float* __restrict__ coef, float* __restrict__ o,
                                  int kcur, float sgn) {
    int i = blockIdx.x * 256 + threadIdx.x;
    int b = i / DD;
    __shared__ float cs[KMAX];
    if (threadIdx.x < kcur) cs[threadIdx.x] = coef[b * KMAX + threadIdx.x];
    __syncthreads();
    const float* ur = Us + (size_t)i * KMAX;
    float acc = 0.f;
    for (int j = 0; j < kcur; ++j) acc += ur[j] * cs[j];
    o[i] = sgn * (acc - v[i]);
}

// denom[b] = sum_d a[b,d]*c[b,d]
__global__ void dot_batch_kernel(const float* __restrict__ a, const float* __restrict__ c,
                                 float* __restrict__ o) {
    int b = blockIdx.x;
    const float* ab = a + (size_t)b * DD;
    const float* cb = c + (size_t)b * DD;
    float s = 0.f;
    for (int d = threadIdx.x; d < DD; d += 256) s += ab[d] * cb[d];
    s = block_reduce_256(s);
    if (threadIdx.x == 0) o[b] = s;
}

// u = (dx - mv)/denom~ ; Us[:,:,k]=u ; VTs[:,k,:]=vT
__global__ void write_uv_kernel(const float* __restrict__ dx, const float* __restrict__ mv,
                                const float* __restrict__ vT, const float* __restrict__ denom,
                                float* __restrict__ Us, float* __restrict__ VTs, int k) {
    int i = blockIdx.x * 256 + threadIdx.x;
    int b = i / DD, d = i % DD;
    float den = denom[b];
    if (fabsf(den) < 1e-9f) den = 1e-9f;
    Us[(size_t)i * KMAX + k] = (dx[i] - mv[i]) / den;
    VTs[((size_t)b * KMAX + k) * DD + d] = vT[i];
}

// two-stage squared-norm reduction; stage2 also does best-objective bookkeeping
__global__ void sq_part_kernel(const float* __restrict__ a, float* __restrict__ part, int n) {
    float s = 0.f;
    for (int i = blockIdx.x * 256 + threadIdx.x; i < n; i += 256 * gridDim.x) s += a[i] * a[i];
    s = block_reduce_256(s);
    if (threadIdx.x == 0) part[blockIdx.x] = s;
}
// mode 0: init best (scal[1]=sum, flag=0). mode 1: flag = sum<best; best=min.
__global__ void sum_best_kernel(const float* __restrict__ part, int n,
                                float* __restrict__ scal, int mode) {
    float s = 0.f;
    for (int i = threadIdx.x; i < n; i += 256) s += part[i];
    s = block_reduce_256(s);
    if (threadIdx.x == 0) {
        scal[0] = s;
        if (mode == 0) {
            scal[1] = s;
            ((int*)scal)[2] = 0;
        } else {
            int f = s < scal[1];
            if (f) scal[1] = s;
            ((int*)scal)[2] = f;
        }
    }
}
__global__ void sel_best_kernel(const float* __restrict__ x, float* __restrict__ best,
                                const float* __restrict__ scal, int n) {
    int i = blockIdx.x * 256 + threadIdx.x;
    if (i < n && ((const int*)scal)[2]) best[i] = x[i];
}

// ---------------- host orchestration ----------------
extern "C" void kernel_launch(void* const* d_in, const int* in_sizes, int n_in,
                              void* d_out, int out_size, void* d_ws, size_t ws_size,
                              hipStream_t stream) {
    const float* x = (const float*)d_in[0];
    const float* A = (const float*)d_in[1];
    const float* U = (const float*)d_in[2];
    const float* b = (const float*)d_in[3];
    float* out = (float*)d_out;

    char* p = (char*)d_ws;
    auto alloc = [&](size_t bytes) -> void* {
        void* r = (void*)p;
        p += (bytes + 255) & ~(size_t)255;
        return r;
    };
    uint16_t* zp = (uint16_t*)alloc((size_t)BB * COUT * PADP * 2);  // padded bf16 iterate
    uint16_t* xp = (uint16_t*)alloc((size_t)BB * CIN * PADP * 2);   // padded bf16 input
    uint16_t* Ah = (uint16_t*)alloc((size_t)COUT * 9 * COUT * 2);   // fragment-packed A
    uint16_t* Uh = (uint16_t*)alloc((size_t)CIN * 9 * COUT * 2);    // fragment-packed U
    float* ux    = (float*)alloc((size_t)NT * 4);
    float* zA    = (float*)alloc((size_t)NT * 4);   // x iterate
    float* gxA   = (float*)alloc((size_t)NT * 4);
    float* zBuf  = (float*)alloc((size_t)NT * 4);   // x_new
    float* gxB   = (float*)alloc((size_t)NT * 4);
    float* upd   = (float*)alloc((size_t)NT * 4);   // update (== dx)
    float* dgx   = (float*)alloc((size_t)NT * 4);
    float* vT    = (float*)alloc((size_t)NT * 4);
    float* mv    = (float*)alloc((size_t)NT * 4);
    float* best  = (float*)alloc((size_t)NT * 4);
    float* Us    = (float*)alloc((size_t)NT * KMAX * 4);
    float* VTs   = (float*)alloc((size_t)NT * KMAX * 4);
    float* t1    = (float*)alloc((size_t)BB * KMAX * 4);
    float* sK    = (float*)alloc((size_t)BB * KMAX * 4);
    float* denom = (float*)alloc((size_t)BB * 4);
    float* part  = (float*)alloc(1024 * 4);
    float* scal  = (float*)alloc(16 * 4);           // [0]=obj [1]=best [2]=flag

    const int EB = NT / 256;                        // elementwise grid (8192)
    const int CONVB = NPOS / 64;                    // conv grid (256)

    // ---- setup ----
    (void)hipMemsetAsync(zA, 0, (size_t)NT * 4, stream);
    (void)hipMemsetAsync(best, 0, (size_t)NT * 4, stream);
    (void)hipMemsetAsync(zp, 0, (size_t)BB * COUT * PADP * 2, stream);  // borders stay zero
    (void)hipMemsetAsync(xp, 0, (size_t)BB * CIN * PADP * 2, stream);
    pack_w_frag_kernel<<<(COUT * COUT * 9 + 255) / 256, 256, 0, stream>>>(A, COUT, Ah);
    pack_w_frag_kernel<<<(COUT * CIN * 9 + 255) / 256, 256, 0, stream>>>(U, CIN, Uh);
    pad_cvt_kernel<<<(BB * CIN * HH * WW) / 256, 256, 0, stream>>>(x, xp);
    // ux = conv(x,U)+b
    conv_wmma_kernel<CIN, 0><<<CONVB, 256, 0, stream>>>(xp, Uh, b, nullptr, nullptr, ux, nullptr);
    // gx = g(0) = relu(ux)   (zp interior is zero already)
    conv_wmma_kernel<COUT, 1><<<CONVB, 256, 0, stream>>>(zp, Ah, ux, zA, nullptr, gxA, nullptr);
    // best_obj = ||gx||^2
    sq_part_kernel<<<1024, 256, 0, stream>>>(gxA, part, NT);
    sum_best_kernel<<<1, 256, 0, stream>>>(part, 1024, scal, 0);
    // update = gx
    lowrank_mv_kernel<<<EB, 256, 0, stream>>>(gxA, Us, sK, upd, 0, -1.f);

    float* xcur = zA;   float* gxcur = gxA;
    float* xnew = zBuf; float* gxnew = gxB;

    for (int k = 0; k < KMAX; ++k) {
        // x_new = x + update (also emits padded bf16); gx_new & dgx fused into conv
        vadd_pad_kernel<<<EB, 256, 0, stream>>>(xcur, upd, xnew, zp);
        conv_wmma_kernel<COUT, 3><<<CONVB, 256, 0, stream>>>(zp, Ah, ux, xnew, gxcur, gxnew, dgx);
        if (k > 0) colsdot_kernel<<<BB * k, 256, 0, stream>>>(upd, Us, t1, k);
        combine_rows_kernel<<<EB, 256, 0, stream>>>(upd, t1, VTs, vT, k);
        if (k > 0) rowsdot_kernel<<<BB * k, 256, 0, stream>>>(dgx, VTs, sK, k);
        lowrank_mv_kernel<<<EB, 256, 0, stream>>>(dgx, Us, sK, mv, k, 1.f);
        dot_batch_kernel<<<BB, 256, 0, stream>>>(vT, dgx, denom);
        write_uv_kernel<<<EB, 256, 0, stream>>>(upd, mv, vT, denom, Us, VTs, k);
        float* t;
        t = xcur; xcur = xnew; xnew = t;
        t = gxcur; gxcur = gxnew; gxnew = t;
        sq_part_kernel<<<1024, 256, 0, stream>>>(gxcur, part, NT);
        sum_best_kernel<<<1, 256, 0, stream>>>(part, 1024, scal, 1);
        sel_best_kernel<<<EB, 256, 0, stream>>>(xcur, best, scal, NT);
        rowsdot_kernel<<<BB * (k + 1), 256, 0, stream>>>(gxcur, VTs, sK, k + 1);
        lowrank_mv_kernel<<<EB, 256, 0, stream>>>(gxcur, Us, sK, upd, k + 1, -1.f);
    }

    // final: out = relu(conv(best, A) + ux)
    pad_cvt_kernel<<<EB, 256, 0, stream>>>(best, zp);
    conv_wmma_kernel<COUT, 2><<<CONVB, 256, 0, stream>>>(zp, Ah, ux, nullptr, nullptr, out, nullptr);
}
